// GATLayer_57896159150204
// MI455X (gfx1250) — compile-verified, matrix operand order checked by reading, not run on hardware
//
#include <hip/hip_runtime.h>
#include <stdint.h>
#include <math.h>

typedef __attribute__((ext_vector_type(16))) _Float16 v16h;
typedef __attribute__((ext_vector_type(8)))  float    v8f;

#define V_NODES 100000
#define E_EDGES 800000
#define D 128
#define H_HEADS 4
#define LEAK 0.2f
#define RSQRT_FH 0.17677669529663687f  // 1/sqrt(32)

// ---------------- conversion / packing ----------------

__global__ void __launch_bounds__(256) k_cvt_f16(const float* __restrict__ src,
                                                 _Float16* __restrict__ dst, int n) {
    int i = blockIdx.x * 256 + threadIdx.x;
    if (i < n) dst[i] = (_Float16)src[i];
}

__global__ void __launch_bounds__(256) k_fill0_i(int* __restrict__ p, int n) {
    int i = blockIdx.x * 256 + threadIdx.x;
    if (i < n) p[i] = 0;
}

// Pack W[128x128] (row-major [k][n], f32) into B-fragment layout:
// frag[kt(0..3)][nt(0..7)][lane(0..31)][v(0..7)] : uint32 = (f16 W[k][n], f16 W[k+1][n])
__global__ void __launch_bounds__(256) k_pack_w(const float* __restrict__ W,
                                                uint32_t* __restrict__ frag) {
    int idx = blockIdx.x * 256 + threadIdx.x;   // 4*8*32*8 = 8192
    if (idx >= 8192) return;
    int v = idx & 7, lane = (idx >> 3) & 31, nt = (idx >> 8) & 7, kt = (idx >> 11) & 3;
    int n = nt * 16 + (lane & 15);
    int khalf = lane >> 4;
    int k = kt * 32 + (v & 3) * 2 + khalf * 8 + (v >> 2) * 16;
    union { _Float16 h[2]; uint32_t u; } pk;
    pk.h[0] = (_Float16)W[k * D + n];
    pk.h[1] = (_Float16)W[(k + 1) * D + n];
    frag[idx] = pk.u;
}

// ---------------- WMMA fragment helpers ----------------

__device__ __forceinline__ v16h load_a_frag(const _Float16* __restrict__ a16,
                                            int row0, int ks) {
    int lane = threadIdx.x & 31;
    int m = lane & 15, khalf = lane >> 4;
    union { v16h v; uint32_t u[8]; } r;
    const uint32_t* base = (const uint32_t*)(a16 + (size_t)(row0 + m) * D);
#pragma unroll
    for (int v = 0; v < 8; ++v) {
        int k = ks * 32 + (v & 3) * 2 + khalf * 8 + (v >> 2) * 16;
        r.u[v] = base[k >> 1];
    }
    return r.v;
}

__device__ __forceinline__ v16h load_b_frag(const uint32_t* __restrict__ frag,
                                            int ks, int nt) {
    int lane = threadIdx.x & 31;
    return *(const v16h*)(frag + ((size_t)(ks * 8 + nt) * 32 + lane) * 8);
}

// ---------------- fused Q/K/V projection GEMM ----------------
__global__ void __launch_bounds__(128) k_gemm_qkv(
    const _Float16* __restrict__ h16,
    const uint32_t* __restrict__ fq, const uint32_t* __restrict__ fk,
    const uint32_t* __restrict__ fv,
    const float* __restrict__ bq, const float* __restrict__ bk,
    const float* __restrict__ bv,
    float* __restrict__ Q, float* __restrict__ K, float* __restrict__ Vv) {
    int wave = threadIdx.x >> 5, lane = threadIdx.x & 31;
    int row0 = blockIdx.x * 16;
    v8f acc[3][2];
#pragma unroll
    for (int m = 0; m < 3; ++m)
#pragma unroll
        for (int t = 0; t < 2; ++t)
#pragma unroll
            for (int i = 0; i < 8; ++i) acc[m][t][i] = 0.0f;

#pragma unroll
    for (int ks = 0; ks < 4; ++ks) {
        v16h a = load_a_frag(h16, row0, ks);
#pragma unroll
        for (int t = 0; t < 2; ++t) {
            int nt = wave * 2 + t;
            v16h b0 = load_b_frag(fq, ks, nt);
            acc[0][t] = __builtin_amdgcn_wmma_f32_16x16x32_f16(
                false, a, false, b0, (short)0, acc[0][t], false, false);
            v16h b1 = load_b_frag(fk, ks, nt);
            acc[1][t] = __builtin_amdgcn_wmma_f32_16x16x32_f16(
                false, a, false, b1, (short)0, acc[1][t], false, false);
            v16h b2 = load_b_frag(fv, ks, nt);
            acc[2][t] = __builtin_amdgcn_wmma_f32_16x16x32_f16(
                false, a, false, b2, (short)0, acc[2][t], false, false);
        }
    }
    int n = lane & 15, mbase = (lane >> 4) * 8;
#pragma unroll
    for (int t = 0; t < 2; ++t) {
        int col = (wave * 2 + t) * 16 + n;
        float cq = bq[col], ck = bk[col], cv = bv[col];
#pragma unroll
        for (int v = 0; v < 8; ++v) {
            size_t o = (size_t)(row0 + mbase + v) * D + col;
            Q[o]  = acc[0][t][v] + cq;
            K[o]  = acc[1][t][v] + ck;
            Vv[o] = acc[2][t][v] + cv;
        }
    }
}

// ---------------- CSR build: count / scan / scatter ----------------

__global__ void __launch_bounds__(256) k_count(const int* __restrict__ ei,
                                               int* __restrict__ cnt) {
    int e = blockIdx.x * 256 + threadIdx.x;
    if (e < E_EDGES) atomicAdd(&cnt[ei[E_EDGES + e]], 1);
}

// per-block exclusive scan (256 elems), writes block total
__global__ void __launch_bounds__(256) k_scan_block(const int* __restrict__ cnt,
                                                    int* __restrict__ excl,
                                                    int* __restrict__ bsum) {
    __shared__ int buf[256];
    int i = blockIdx.x * 256 + threadIdx.x;
    int x = (i < V_NODES) ? cnt[i] : 0;
    buf[threadIdx.x] = x;
    __syncthreads();
    for (int off = 1; off < 256; off <<= 1) {
        int t = (threadIdx.x >= off) ? buf[threadIdx.x - off] : 0;
        __syncthreads();
        buf[threadIdx.x] += t;
        __syncthreads();
    }
    if (i < V_NODES) excl[i] = buf[threadIdx.x] - x;
    if (threadIdx.x == 255) bsum[blockIdx.x] = buf[255];
}

// single-block exclusive scan of block sums (nb <= 512)
__global__ void __launch_bounds__(512) k_scan_bsum(int* __restrict__ bsum, int nb) {
    __shared__ int buf[512];
    int x = (threadIdx.x < nb) ? bsum[threadIdx.x] : 0;
    buf[threadIdx.x] = x;
    __syncthreads();
    for (int off = 1; off < 512; off <<= 1) {
        int t = (threadIdx.x >= off) ? buf[threadIdx.x - off] : 0;
        __syncthreads();
        buf[threadIdx.x] += t;
        __syncthreads();
    }
    if (threadIdx.x < nb) bsum[threadIdx.x] = buf[threadIdx.x] - x;
}

__global__ void __launch_bounds__(256) k_scan_add(int* __restrict__ row_start,
                                                  const int* __restrict__ bsum,
                                                  int* __restrict__ cursor) {
    int i = blockIdx.x * 256 + threadIdx.x;
    if (i < V_NODES) {
        int r = row_start[i] + bsum[blockIdx.x];
        row_start[i] = r;
        cursor[i] = r;
    }
}

__global__ void __launch_bounds__(256) k_scatter(const int* __restrict__ ei,
                                                 int* __restrict__ cursor,
                                                 int* __restrict__ elist) {
    int e = blockIdx.x * 256 + threadIdx.x;
    if (e < E_EDGES) {
        int pos = atomicAdd(&cursor[ei[E_EDGES + e]], 1);
        elist[pos] = e;
    }
}

// ---------------- edge pass: un-normalized attention scores ----------------
// 1 wave per edge, 8 waves per block; coalesced 512B loads of Q[src], K[tgt].
__global__ void __launch_bounds__(256) k_edge_attn(
    const int* __restrict__ ei, const float* __restrict__ ew,
    const float* __restrict__ Q, const float* __restrict__ K,
    const float* __restrict__ We, const float* __restrict__ be,
    float* __restrict__ attn) {
    int wave = threadIdx.x >> 5, lane = threadIdx.x & 31;
    long e = (long)blockIdx.x * 8 + wave;
    if (e >= E_EDGES) return;
    int src = ei[e], tgt = ei[E_EDGES + e];
    float4 q = ((const float4*)(Q + (size_t)src * D))[lane];
    float4 k = ((const float4*)(K + (size_t)tgt * D))[lane];
    float p = q.x * k.x + q.y * k.y + q.z * k.z + q.w * k.w;
    p += __shfl_xor(p, 1, 32);
    p += __shfl_xor(p, 2, 32);
    p += __shfl_xor(p, 4, 32);
    if ((lane & 7) == 0) {
        int head = lane >> 3;
        float w = ew[e];
        float eb = w * We[head] + be[head];
        eb = eb >= 0.0f ? eb : LEAK * eb;
        attn[e * H_HEADS + head] = expf(p * RSQRT_FH + eb);
    }
}

// ---------------- per-target gather aggregation (no float atomics) ----------
// 1 wave per target node: sum attn-exp (softmax denom), then FMA alpha*omega*V[src]
// into registers; single coalesced 512B store per node.
__global__ void __launch_bounds__(256) k_target_aggr(
    const int* __restrict__ ei, const float* __restrict__ ew,
    const int* __restrict__ row_start, const int* __restrict__ cnt,
    const int* __restrict__ elist,
    const float* __restrict__ Vv, const float* __restrict__ attn,
    float* __restrict__ aggr) {
    int wave = threadIdx.x >> 5, lane = threadIdx.x & 31;
    int v = blockIdx.x * 8 + wave;
    if (v >= V_NODES) return;
    int rs = row_start[v], n = cnt[v];
    int head = lane >> 3;

    float degsum = 0.0f;
    for (int j = 0; j < n; ++j) {
        int e = elist[rs + j];
        degsum += attn[(size_t)e * H_HEADS + head];
    }
    float inv = 1.0f / (degsum + 1e-16f);

    float4 acc = make_float4(0.f, 0.f, 0.f, 0.f);
    for (int j = 0; j < n; ++j) {
        int e = elist[rs + j];
        float alpha = attn[(size_t)e * H_HEADS + head] * inv * ew[e];
        int src = ei[e];
        float4 x = ((const float4*)(Vv + (size_t)src * D))[lane];
        acc.x += x.x * alpha;
        acc.y += x.y * alpha;
        acc.z += x.z * alpha;
        acc.w += x.w * alpha;
    }
    ((float4*)(aggr + (size_t)v * D))[lane] = acc;
}

// ---------------- output GEMM + residual + LayerNorm + LeakyReLU ------------
__global__ void __launch_bounds__(128) k_gemm_o_ln(
    const _Float16* __restrict__ a16, const uint32_t* __restrict__ fo,
    const float* __restrict__ bo, const float* __restrict__ hres,
    const float* __restrict__ gamma, const float* __restrict__ beta,
    float* __restrict__ out) {
    __shared__ float tile[16][D + 4];
    __shared__ float mu_s[16], rs_s[16];
    int wave = threadIdx.x >> 5, lane = threadIdx.x & 31;
    int row0 = blockIdx.x * 16;
    v8f acc[2];
#pragma unroll
    for (int t = 0; t < 2; ++t)
#pragma unroll
        for (int i = 0; i < 8; ++i) acc[t][i] = 0.0f;

#pragma unroll
    for (int ks = 0; ks < 4; ++ks) {
        v16h a = load_a_frag(a16, row0, ks);
#pragma unroll
        for (int t = 0; t < 2; ++t) {
            v16h b = load_b_frag(fo, ks, wave * 2 + t);
            acc[t] = __builtin_amdgcn_wmma_f32_16x16x32_f16(
                false, a, false, b, (short)0, acc[t], false, false);
        }
    }
    int n = lane & 15, mbase = (lane >> 4) * 8;
#pragma unroll
    for (int t = 0; t < 2; ++t) {
        int col = (wave * 2 + t) * 16 + n;
        float c = bo[col];
#pragma unroll
        for (int v = 0; v < 8; ++v) {
            int r = mbase + v;
            tile[r][col] = acc[t][v] + c + hres[(size_t)(row0 + r) * D + col];
        }
    }
    __syncthreads();
    if (threadIdx.x < 16) {
        int r = threadIdx.x;
        float s = 0.0f, s2 = 0.0f;
#pragma unroll 4
        for (int c = 0; c < D; ++c) {
            float x = tile[r][c];
            s += x; s2 += x * x;
        }
        float mu = s * (1.0f / D);
        float var = s2 * (1.0f / D) - mu * mu;
        mu_s[r] = mu;
        rs_s[r] = rsqrtf(var + 1e-5f);
    }
    __syncthreads();
    for (int i = threadIdx.x; i < 16 * D; i += 128) {
        int r = i >> 7, c = i & (D - 1);
        float x = (tile[r][c] - mu_s[r]) * rs_s[r] * gamma[c] + beta[c];
        out[(size_t)(row0 + r) * D + c] = x >= 0.0f ? x : LEAK * x;
    }
}

// ---------------- host ----------------

extern "C" void kernel_launch(void* const* d_in, const int* in_sizes, int n_in,
                              void* d_out, int out_size, void* d_ws, size_t ws_size,
                              hipStream_t stream) {
    const float* h   = (const float*)d_in[0];
    const int*   ei  = (const int*)d_in[1];
    const float* ew  = (const float*)d_in[2];
    const float* Wq  = (const float*)d_in[3];
    const float* bq  = (const float*)d_in[4];
    const float* Wk  = (const float*)d_in[5];
    const float* bk  = (const float*)d_in[6];
    const float* Wv  = (const float*)d_in[7];
    const float* bv  = (const float*)d_in[8];
    const float* Wo  = (const float*)d_in[9];
    const float* bo  = (const float*)d_in[10];
    const float* We  = (const float*)d_in[11];
    const float* be  = (const float*)d_in[12];
    const float* gamma = (const float*)d_in[13];
    const float* beta  = (const float*)d_in[14];
    float* out = (float*)d_out;

    size_t off = 0;
    char* base = (char*)d_ws;
    auto alloc = [&](size_t bytes) -> void* {
        void* p = base + off;
        off += (bytes + 255) & ~(size_t)255;
        return p;
    };
    const size_t VD = (size_t)V_NODES * D;
    float*     Q    = (float*)alloc(VD * sizeof(float));
    float*     K    = (float*)alloc(VD * sizeof(float));
    float*     Vv   = (float*)alloc(VD * sizeof(float));
    _Float16*  h16  = (_Float16*)alloc(VD * sizeof(_Float16));
    _Float16*  a16  = (_Float16*)alloc(VD * sizeof(_Float16));
    float*     attn = (float*)alloc((size_t)E_EDGES * H_HEADS * sizeof(float));
    int*       cnt  = (int*)alloc(V_NODES * sizeof(int));
    int*       rowS = (int*)alloc(V_NODES * sizeof(int));
    int*       curs = (int*)alloc(V_NODES * sizeof(int));
    int*       elist= (int*)alloc(E_EDGES * sizeof(int));
    int*       bsum = (int*)alloc(512 * sizeof(int));
    uint32_t*  fq   = (uint32_t*)alloc(8192 * 4);
    uint32_t*  fk   = (uint32_t*)alloc(8192 * 4);
    uint32_t*  fv   = (uint32_t*)alloc(8192 * 4);
    uint32_t*  fo   = (uint32_t*)alloc(8192 * 4);
    float*     aggr = Q;  // Q dead after k_edge_attn; reuse as aggregation buffer

    const int cvt_blocks  = (int)(VD / 256);          // 50000
    const int mtiles      = V_NODES / 16;             // 6250
    const int edge_waves  = E_EDGES / 8;              // 100000 (wave-per-edge)
    const int edge_thr    = (E_EDGES + 255) / 256;    // 3125   (thread-per-edge)
    const int node_thr    = (V_NODES + 255) / 256;    // 391
    const int node_waves  = (V_NODES + 7) / 8;        // 12500  (wave-per-node)

    // projections
    k_cvt_f16<<<cvt_blocks, 256, 0, stream>>>(h, h16, (int)VD);
    k_pack_w<<<32, 256, 0, stream>>>(Wq, fq);
    k_pack_w<<<32, 256, 0, stream>>>(Wk, fk);
    k_pack_w<<<32, 256, 0, stream>>>(Wv, fv);
    k_pack_w<<<32, 256, 0, stream>>>(Wo, fo);
    k_gemm_qkv<<<mtiles, 128, 0, stream>>>(h16, fq, fk, fv, bq, bk, bv, Q, K, Vv);

    // CSR of incoming edges per target (integer atomics only)
    k_fill0_i<<<node_thr, 256, 0, stream>>>(cnt, V_NODES);
    k_count<<<edge_thr, 256, 0, stream>>>(ei, cnt);
    k_scan_block<<<node_thr, 256, 0, stream>>>(cnt, rowS, bsum);
    k_scan_bsum<<<1, 512, 0, stream>>>(bsum, node_thr);
    k_scan_add<<<node_thr, 256, 0, stream>>>(rowS, bsum, curs);
    k_scatter<<<edge_thr, 256, 0, stream>>>(ei, curs, elist);

    // attention + gather-style aggregation (no float atomics anywhere)
    k_edge_attn<<<edge_waves, 256, 0, stream>>>(ei, ew, Q, K, We, be, attn);
    k_target_aggr<<<node_waves, 256, 0, stream>>>(ei, ew, rowS, cnt, elist, Vv, attn, aggr);

    // output projection + residual + LN + LeakyReLU
    k_cvt_f16<<<cvt_blocks, 256, 0, stream>>>(aggr, a16, (int)VD);
    k_gemm_o_ln<<<mtiles, 128, 0, stream>>>(a16, fo, bo, h, gamma, beta, out);
}
